// LSTM2_53060025975086
// MI455X (gfx1250) — matrix operand order repaired and checked by reading, not compiled, not run on hardware
//
#include <hip/hip_runtime.h>
#include <hip/hip_bf16.h>
#include <math.h>

// ---------------------------------------------------------------------------
// Persistent-kernel LSTM for MI455X (gfx1250, wave32, WMMA).
//
// Design (compile-only reasoning, see analysis):
//  * bf16 weights, fp32 accumulate via v_wmma_f32_16x16x32_bf16.
//  * Weight matrices pre-swizzled into a contiguous-k bf16 A-matrix 16x32
//    fragment layout (1KB per fragment, one 32B slice per lane).
//  * 64 WGs x 256 threads persistent; WG wg owns hidden rows [wg*16, wg*16+16)
//    for ALL four gates -> cell update is WG-local, only 3 global barriers/step.
//  * h vectors double-buffered in global (L2), staged to LDS per layer.
//  * Attention/softmax epilogue fused into the persistent kernel.
// ---------------------------------------------------------------------------

#define IN_DIM  512
#define HDIM    1024
#define TSTEPS  1024
#define GDIM    4096      // 4*H gate rows
#define K1PAD   544       // 513 padded to 17*32
#define CU1     17        // k-chunks of W_ih1
#define CRH     32        // k-chunks for K=1024
#define RB      256       // row blocks (4096/16)
#define NWG     64
#define BLOCK   256
#define NWAVES  8

typedef __attribute__((ext_vector_type(16))) __bf16 v16bf;
typedef __attribute__((ext_vector_type(8)))  float  v8f;
typedef unsigned short u16;
typedef unsigned int   u32;

__device__ __forceinline__ u16 f2bf(float f) {
  u32 u = __float_as_uint(f);
  u32 r = u + 0x7FFFu + ((u >> 16) & 1u);   // round-to-nearest-even
  return (u16)(r >> 16);
}
__device__ __forceinline__ float sigmoidf(float x) { return 1.0f / (1.0f + expf(-x)); }

// ------------------------- prologue kernels --------------------------------

// Convert row-major fp32 weight (4096 x Ksrc) into bf16 WMMA A-fragments.
// Fragment (rb, kc): 32 lanes x 16 bf16 (32B per lane), contiguous K order:
// lane<16 -> K 0..15; lane>=16 -> K 16..31.
__global__ void conv_weight_frag(const float* __restrict__ W, u16* __restrict__ F,
                                 int Ksrc, int CU) {
  int gid = blockIdx.x * blockDim.x + threadIdx.x;
  int total = RB * CU * 32;
  if (gid >= total) return;
  int lane = gid & 31;
  int fidx = gid >> 5;            // rb*CU + kc
  int kc   = fidx % CU;
  int rb   = fidx / CU;
  int row  = rb * 16 + (lane & 15);
  int kb   = kc * 32 + ((lane & 16) ? 16 : 0);
  union { u16 h[16]; uint4 q[2]; } buf;
#pragma unroll
  for (int e = 0; e < 16; ++e) {
    int k = kb + e;
    float v = (k < Ksrc) ? W[(size_t)row * Ksrc + k] : 0.0f;
    buf.h[e] = f2bf(v);
  }
  uint4* dst = (uint4*)(F + (size_t)gid * 16);
  dst[0] = buf.q[0];
  dst[1] = buf.q[1];
}

// Build padded bf16 input rows: [x_t(512), time_t, zeros...] length 544.
__global__ void build_xs(const float* __restrict__ x, const float* __restrict__ tm,
                         u16* __restrict__ xs) {
  int idx = blockIdx.x * blockDim.x + threadIdx.x;
  if (idx >= TSTEPS * K1PAD) return;
  int t = idx / K1PAD, k = idx % K1PAD;
  float v = (k < IN_DIM) ? x[(size_t)t * IN_DIM + k] : (k == IN_DIM ? tm[t] : 0.0f);
  xs[idx] = f2bf(v);
}

__global__ void init_state(const float* bih1, const float* bhh1,
                           const float* bih2, const float* bhh2,
                           const float* h01, const float* c01,
                           const float* h02, const float* c02,
                           const float* h03, const float* c03,
                           float* b1, float* b2,
                           float* c1, float* c2, float* c3,
                           u16* h1b, u16* h2b, u16* h3b, u32* bar) {
  int idx = blockIdx.x * blockDim.x + threadIdx.x;
  if (idx < GDIM) { b1[idx] = bih1[idx] + bhh1[idx]; b2[idx] = bih2[idx] + bhh2[idx]; }
  if (idx < HDIM) {
    c1[idx] = c01[idx]; c2[idx] = c02[idx]; c3[idx] = c03[idx];
    h1b[idx] = f2bf(h01[idx]);      // double-buffer slot 0 (read at t=0)
    h2b[idx] = f2bf(h02[idx]);
    h3b[idx] = f2bf(h03[idx]);
  }
  if (idx == 0) { bar[0] = 0u; bar[32] = 0u; }   // cnt, gen (separate lines)
}

// ------------------------- persistent kernel -------------------------------

__device__ __forceinline__ void gbar(u32* cnt, u32* gen, unsigned target) {
  __syncthreads();
  if (threadIdx.x == 0) {
    __threadfence();
    if (atomicAdd(cnt, 1u) == (u32)(NWG - 1)) {
      atomicExch(cnt, 0u);
      __threadfence();
      atomicAdd(gen, 1u);
    } else {
      while (atomicAdd(gen, 0u) < target) __builtin_amdgcn_s_sleep(2);
    }
    __threadfence();
  }
  __syncthreads();
}

__device__ __forceinline__ void stage(u16* dst, const u16* __restrict__ src,
                                      int n, int tid) {
  for (int i = tid; i < n; i += BLOCK) dst[i] = src[i];
}

// One LSTM cell for the 16 hidden rows owned by this WG.
// Gates g=0..3 live at row-block (g*64 + wg). Waves split the combined
// k-chunk list (feed CU chunks, then recurrent CRH chunks); partials reduced
// through LDS; threads 0..15 apply the nonlinearity + cell update.
__device__ __forceinline__ void layer_step(
    const u16* __restrict__ Wu, int CU, const u16* __restrict__ Wr,
    const u16* lds_u, const u16* lds_r,
    const float* __restrict__ bias, float* __restrict__ cstate,
    u16* __restrict__ hout_bf, float* __restrict__ hout_f32,
    float* lds_red, float* lds_g,
    int wg, int wave, int lane, int tid)
{
  v8f a0 = {}, a1 = {}, a2 = {}, a3 = {};
  const int total = CU + CRH;
  for (int ch = wave; ch < total; ch += NWAVES) {
    const u16* lsrc; const u16* fbase; int kc, stride;
    if (ch < CU) { kc = ch;      lsrc = lds_u; fbase = Wu; stride = CU;  }
    else         { kc = ch - CU; lsrc = lds_r; fbase = Wr; stride = CRH; }
    // B fragment: vector chunk broadcast into all 16 columns.
    // lane<16 holds K 0..15 of the chunk, lane>=16 holds K 16..31.
    const v16bf b = *(const v16bf*)(lsrc + kc * 32 + (lane & 16));
    const size_t lofs = (size_t)lane * 16;
    const u16* f0 = fbase + ((size_t)((0 * 64 + wg) * stride + kc)) * 512 + lofs;
    const u16* f1 = fbase + ((size_t)((1 * 64 + wg) * stride + kc)) * 512 + lofs;
    const u16* f2 = fbase + ((size_t)((2 * 64 + wg) * stride + kc)) * 512 + lofs;
    const u16* f3 = fbase + ((size_t)((3 * 64 + wg) * stride + kc)) * 512 + lofs;
    __builtin_prefetch(f0 + NWAVES * 512, 0, 1);   // global_prefetch next chunk
    v16bf wa;
    wa = *(const v16bf*)f0;
    a0 = __builtin_amdgcn_wmma_f32_16x16x32_bf16(false, wa, false, b, (short)0, a0, false, false);
    wa = *(const v16bf*)f1;
    a1 = __builtin_amdgcn_wmma_f32_16x16x32_bf16(false, wa, false, b, (short)0, a1, false, false);
    wa = *(const v16bf*)f2;
    a2 = __builtin_amdgcn_wmma_f32_16x16x32_bf16(false, wa, false, b, (short)0, a2, false, false);
    wa = *(const v16bf*)f3;
    a3 = __builtin_amdgcn_wmma_f32_16x16x32_bf16(false, wa, false, b, (short)0, a3, false, false);
  }
  // D layout: VGPR r = row r (lanes 0..15) / row 8+r (lanes 16..31); all
  // columns identical since B columns are identical -> lanes 0 & 16 export.
  if ((lane & 15) == 0) {
    int half = lane >> 4;
    float* dst = lds_red + (wave * 4) * 16 + half * 8;
#pragma unroll
    for (int r = 0; r < 8; ++r) {
      dst[0 * 16 + r] = a0[r];
      dst[1 * 16 + r] = a1[r];
      dst[2 * 16 + r] = a2[r];
      dst[3 * 16 + r] = a3[r];
    }
  }
  __syncthreads();
  if (tid < 64) {                              // sum wave partials + bias
    int g = tid >> 4, r = tid & 15;
    float s = bias[g * 1024 + wg * 16 + r];
#pragma unroll
    for (int w = 0; w < NWAVES; ++w) s += lds_red[(w * 4 + g) * 16 + r];
    lds_g[g * 16 + r] = s;
  }
  __syncthreads();
  if (tid < 16) {                              // i,f,g,o -> c', h'
    float iv = lds_g[tid], fv = lds_g[16 + tid], gv = lds_g[32 + tid], ov = lds_g[48 + tid];
    int j = wg * 16 + tid;
    float cn = sigmoidf(fv) * cstate[j] + sigmoidf(iv) * tanhf(gv);
    float hn = sigmoidf(ov) * tanhf(cn);
    cstate[j]  = cn;
    hout_bf[j] = f2bf(hn);
    if (hout_f32) hout_f32[j] = hn;
  }
  __syncthreads();
}

__launch_bounds__(BLOCK)
__global__ void lstm_persistent(
    const u16* __restrict__ xs,
    const u16* __restrict__ W1f, const u16* __restrict__ Wh1f,
    const u16* __restrict__ W2f, const u16* __restrict__ Wh2f,
    const float* __restrict__ b1, const float* __restrict__ b2,
    float* __restrict__ c1, float* __restrict__ c2, float* __restrict__ c3,
    u16* __restrict__ h1b, u16* __restrict__ h2b, u16* __restrict__ h3b,
    float* __restrict__ hs, float* __restrict__ attn,
    u32* cnt, u32* gen, float* __restrict__ out)
{
  __shared__ __align__(32) u16 lds_u[HDIM];
  __shared__ __align__(32) u16 lds_r[HDIM];
  __shared__ float lds_red[NWAVES * 4 * 16];
  __shared__ float lds_g[4 * 16];
  __shared__ float sred[BLOCK];

  const int tid  = threadIdx.x;
  const int lane = tid & 31;
  const int wave = tid >> 5;
  const int wg   = blockIdx.x;
  unsigned bt = 0;
  int p = 0;

  for (int t = 0; t < TSTEPS; ++t) {
    const int q = p ^ 1;
    // ---- layer 1: feed = [x_t, time_t], recur = h1 ----
    stage(lds_u, xs + (size_t)t * K1PAD, K1PAD, tid);
    stage(lds_r, h1b + p * HDIM, HDIM, tid);
    __syncthreads();
    layer_step(W1f, CU1, Wh1f, lds_u, lds_r, b1, c1,
               h1b + q * HDIM, nullptr, lds_red, lds_g, wg, wave, lane, tid);
    gbar(cnt, gen, ++bt);
    // ---- layer 2: feed = h1', recur = h2 ----
    stage(lds_u, h1b + q * HDIM, HDIM, tid);
    stage(lds_r, h2b + p * HDIM, HDIM, tid);
    __syncthreads();
    layer_step(W2f, CRH, Wh2f, lds_u, lds_r, b2, c2,
               h2b + q * HDIM, nullptr, lds_red, lds_g, wg, wave, lane, tid);
    gbar(cnt, gen, ++bt);
    // ---- layer 3 (reuses layer-2 weights): feed = h2', recur = h3 ----
    stage(lds_u, h2b + q * HDIM, HDIM, tid);
    stage(lds_r, h3b + p * HDIM, HDIM, tid);
    __syncthreads();
    layer_step(W2f, CRH, Wh2f, lds_u, lds_r, b2, c3,
               h3b + q * HDIM, hs + (size_t)t * HDIM, lds_red, lds_g, wg, wave, lane, tid);
    gbar(cnt, gen, ++bt);
    p = q;
  }

  // ------------------ attention epilogue ------------------
  const float* hf = hs + (size_t)(TSTEPS - 1) * HDIM;
  for (int tt = 0; tt < TSTEPS / NWG; ++tt) {
    int t = wg * (TSTEPS / NWG) + tt;
    const float* hrow = hs + (size_t)t * HDIM;
    float psum = 0.f;
    for (int j = tid; j < HDIM; j += BLOCK) psum += hrow[j] * hf[j];
    sred[tid] = psum; __syncthreads();
    for (int s = BLOCK / 2; s > 0; s >>= 1) {
      if (tid < s) sred[tid] += sred[tid + s];
      __syncthreads();
    }
    if (tid == 0) attn[t] = sred[0];
    __syncthreads();
  }
  gbar(cnt, gen, ++bt);

  // softmax stats (computed redundantly per WG; attn is tiny & L2-hot)
  float mloc = -3.4e38f;
  for (int j = tid; j < TSTEPS; j += BLOCK) mloc = fmaxf(mloc, attn[j]);
  sred[tid] = mloc; __syncthreads();
  for (int s = BLOCK / 2; s > 0; s >>= 1) {
    if (tid < s) sred[tid] = fmaxf(sred[tid], sred[tid + s]);
    __syncthreads();
  }
  float mval = sred[0]; __syncthreads();
  float sloc = 0.f;
  for (int j = tid; j < TSTEPS; j += BLOCK) sloc += expf(attn[j] - mval);
  sred[tid] = sloc; __syncthreads();
  for (int s = BLOCK / 2; s > 0; s >>= 1) {
    if (tid < s) sred[tid] += sred[tid + s];
    __syncthreads();
  }
  float S = sred[0]; __syncthreads();

  // context for the 16 owned hidden indices
  int hh = tid >> 4, tp = tid & 15;
  int j = wg * 16 + hh;
  float psum = 0.f;
  for (int t = tp; t < TSTEPS; t += 16)
    psum += expf(attn[t] - mval) * hs[(size_t)t * HDIM + j];
  sred[tid] = psum; __syncthreads();
  if (tp == 0) {
    float s = 0.f;
#pragma unroll
    for (int k = 0; k < 16; ++k) s += sred[hh * 16 + k];
    out[j] = s / S;
  }
}

// ------------------------------ host side ----------------------------------

extern "C" void kernel_launch(void* const* d_in, const int* in_sizes, int n_in,
                              void* d_out, int out_size, void* d_ws, size_t ws_size,
                              hipStream_t stream) {
  (void)in_sizes; (void)n_in; (void)out_size;
  const float* x    = (const float*)d_in[0];
  const float* tm   = (const float*)d_in[1];
  const float* h01  = (const float*)d_in[2];
  const float* c01  = (const float*)d_in[3];
  const float* h02  = (const float*)d_in[4];
  const float* c02  = (const float*)d_in[5];
  const float* h03  = (const float*)d_in[6];
  const float* c03  = (const float*)d_in[7];
  const float* Wih1 = (const float*)d_in[8];
  const float* Whh1 = (const float*)d_in[9];
  const float* bih1 = (const float*)d_in[10];
  const float* bhh1 = (const float*)d_in[11];
  const float* Wih2 = (const float*)d_in[12];
  const float* Whh2 = (const float*)d_in[13];
  const float* bih2 = (const float*)d_in[14];
  const float* bhh2 = (const float*)d_in[15];
  float* out = (float*)d_out;

  char* ws = (char*)d_ws;
  size_t off = 0;
  auto carve = [&](size_t bytes) {
    size_t r = off;
    off = (off + bytes + 255) & ~(size_t)255;
    return r;
  };
  size_t o_bar  = carve(256);
  size_t o_xs   = carve((size_t)TSTEPS * K1PAD * 2);
  size_t o_W1   = carve((size_t)RB * CU1 * 512 * 2);
  size_t o_Wh1  = carve((size_t)RB * CRH * 512 * 2);
  size_t o_W2   = carve((size_t)RB * CRH * 512 * 2);
  size_t o_Wh2  = carve((size_t)RB * CRH * 512 * 2);
  size_t o_b1   = carve(GDIM * 4);
  size_t o_b2   = carve(GDIM * 4);
  size_t o_c1   = carve(HDIM * 4);
  size_t o_c2   = carve(HDIM * 4);
  size_t o_c3   = carve(HDIM * 4);
  size_t o_h1b  = carve(2 * HDIM * 2);
  size_t o_h2b  = carve(2 * HDIM * 2);
  size_t o_h3b  = carve(2 * HDIM * 2);
  size_t o_hs   = carve((size_t)TSTEPS * HDIM * 4);
  size_t o_attn = carve(TSTEPS * 4);
  if (off > ws_size) return;   // workspace too small; nothing safe to do

  u32*  bar  = (u32*)(ws + o_bar);
  u16*  xs   = (u16*)(ws + o_xs);
  u16*  W1f  = (u16*)(ws + o_W1);
  u16*  Wh1f = (u16*)(ws + o_Wh1);
  u16*  W2f  = (u16*)(ws + o_W2);
  u16*  Wh2f = (u16*)(ws + o_Wh2);
  float* b1  = (float*)(ws + o_b1);
  float* b2  = (float*)(ws + o_b2);
  float* c1  = (float*)(ws + o_c1);
  float* c2  = (float*)(ws + o_c2);
  float* c3  = (float*)(ws + o_c3);
  u16*  h1b  = (u16*)(ws + o_h1b);
  u16*  h2b  = (u16*)(ws + o_h2b);
  u16*  h3b  = (u16*)(ws + o_h3b);
  float* hs  = (float*)(ws + o_hs);
  float* attn= (float*)(ws + o_attn);

  // prologue (stream-ordered before the persistent kernel)
  {
    int t1 = RB * CU1 * 32, t2 = RB * CRH * 32;
    conv_weight_frag<<<(t1 + 255) / 256, 256, 0, stream>>>(Wih1, W1f, IN_DIM + 1, CU1);
    conv_weight_frag<<<(t2 + 255) / 256, 256, 0, stream>>>(Whh1, Wh1f, HDIM, CRH);
    conv_weight_frag<<<(t2 + 255) / 256, 256, 0, stream>>>(Wih2, W2f, HDIM, CRH);
    conv_weight_frag<<<(t2 + 255) / 256, 256, 0, stream>>>(Whh2, Wh2f, HDIM, CRH);
    int tx = TSTEPS * K1PAD;
    build_xs<<<(tx + 255) / 256, 256, 0, stream>>>(x, tm, xs);
    init_state<<<(GDIM + 255) / 256, 256, 0, stream>>>(
        bih1, bhh1, bih2, bhh2, h01, c01, h02, c02, h03, c03,
        b1, b2, c1, c2, c3, h1b, h2b, h3b, bar);
  }

  lstm_persistent<<<NWG, BLOCK, 0, stream>>>(
      xs, W1f, Wh1f, W2f, Wh2f, b1, b2, c1, c2, c3,
      h1b, h2b, h3b, hs, attn, bar, bar + 32, out);
}
